// BiLSTM_29094108463341
// MI455X (gfx1250) — matrix operand order repaired and checked by reading, not compiled, not run on hardware
//
#include <hip/hip_runtime.h>
#include <hip/hip_bf16.h>

// ---------------------------------------------------------------------------
// BiLSTM (2-layer, bidirectional), B=32, T=1024, F=H=512, gates 4H=2048.
// bf16 WMMA (v_wmma_f32_16x16x32_bf16) for all matmuls, fp32 accumulation.
// Recurrence: persistent kernel, Wh^T resident in LDS (320KB/WGP), cell
// state in VGPRs, one grid barrier per time step (double-buffered h).
// ---------------------------------------------------------------------------

#define BB   32
#define TT   1024
#define FF   512
#define HH   512
#define NG   2048          // 4*H
#define BT   (BB*TT)       // 32768
#define LPITCH 520         // padded LDS row pitch (bf16 elems): 1040B = 65*16B

typedef __attribute__((ext_vector_type(16))) __bf16 v16bf;
typedef __attribute__((ext_vector_type(8)))  float  v8f;

union BFrag {
    v16bf v;
    uint4 u[2];
};

static __device__ __forceinline__ float sigmoidf_(float x) {
    return 1.0f / (1.0f + __expf(-x));
}

// -------------------------------------------------------------------- convert
__global__ void cvt_f32_to_bf16(const float* __restrict__ in,
                                __bf16* __restrict__ out, int n) {
    for (int i = blockIdx.x * blockDim.x + threadIdx.x; i < n;
         i += gridDim.x * blockDim.x)
        out[i] = (__bf16)in[i];
}

// W: [K, 2048] fp32 row-major  ->  Wt: [2048, K] bf16 row-major
__global__ void transpose_cvt(const float* __restrict__ W,
                              __bf16* __restrict__ Wt, int K) {
    size_t n = (size_t)K * NG;
    for (size_t i = (size_t)blockIdx.x * blockDim.x + threadIdx.x; i < n;
         i += (size_t)gridDim.x * blockDim.x) {
        size_t k = i / NG, c = i % NG;
        Wt[c * (size_t)K + k] = (__bf16)W[i];
    }
}

// ------------------------------------------------------------ projection GEMM
// C[M, 2048] = A[M,K](bf16, row-major) x Wt[2048,K](bf16, pre-transposed) + bias
// Block: 256 threads = 8 waves; block tile 64(M) x 64(N); wave tile 16 x 32.
__global__ __launch_bounds__(256) void gemm_xproj(
    const __bf16* __restrict__ A, const __bf16* __restrict__ Wt,
    const float* __restrict__ bias, float* __restrict__ C,
    int M, int K)
{
    const int lane = threadIdx.x & 31;
    const int wave = threadIdx.x >> 5;
    const int m0 = blockIdx.y * 64 + (wave & 3) * 16;
    const int n0 = blockIdx.x * 64 + (wave >> 2) * 32;
    const int lm   = lane & 15;
    const int hi16 = lane >> 4;                 // 0: lanes 0-15, 1: lanes 16-31

    const __bf16* arow  = A  + (size_t)(m0 + lm)      * K;
    const __bf16* b0row = Wt + (size_t)(n0 + lm)      * K;
    const __bf16* b1row = Wt + (size_t)(n0 + 16 + lm) * K;

    v8f acc0 = {};
    v8f acc1 = {};
    for (int k = 0; k < K; k += 32) {
        BFrag a, b0, b1;
        const int ka = k + hi16 * 16;
        a.u[0] = *(const uint4*)(arow + ka);
        a.u[1] = *(const uint4*)(arow + ka + 8);
        const int kb = k + hi16 * 16;
        b0.u[0] = *(const uint4*)(b0row + kb);
        b0.u[1] = *(const uint4*)(b0row + kb + 8);
        b1.u[0] = *(const uint4*)(b1row + kb);
        b1.u[1] = *(const uint4*)(b1row + kb + 8);
        acc0 = __builtin_amdgcn_wmma_f32_16x16x32_bf16(false, a.v, false, b0.v,
                                                       (short)0, acc0, false, false);
        acc1 = __builtin_amdgcn_wmma_f32_16x16x32_bf16(false, a.v, false, b1.v,
                                                       (short)0, acc1, false, false);
    }
    #pragma unroll
    for (int v = 0; v < 8; ++v) {
        const int m = m0 + v + hi16 * 8;
        float* crow = C + (size_t)m * NG;
        crow[n0 + lm]      = acc0[v] + bias[n0 + lm];
        crow[n0 + 16 + lm] = acc1[v] + bias[n0 + 16 + lm];
    }
}

// ---------------------------------------------------------- grid-wide barrier
// Split-counter barrier among `nb` blocks of one direction.
static __device__ __forceinline__ void gridbar(unsigned* cnt, unsigned* gen,
                                               unsigned nb) {
    __threadfence();
    __syncthreads();
    if (threadIdx.x == 0) {
        unsigned g = __hip_atomic_load(gen, __ATOMIC_RELAXED,
                                       __HIP_MEMORY_SCOPE_AGENT);
        unsigned prev = __hip_atomic_fetch_add(cnt, 1u, __ATOMIC_ACQ_REL,
                                               __HIP_MEMORY_SCOPE_AGENT);
        if (prev == nb - 1u) {
            __hip_atomic_store(cnt, 0u, __ATOMIC_RELAXED,
                               __HIP_MEMORY_SCOPE_AGENT);
            __hip_atomic_store(gen, g + 1u, __ATOMIC_RELEASE,
                               __HIP_MEMORY_SCOPE_AGENT);
        } else {
            while (__hip_atomic_load(gen, __ATOMIC_ACQUIRE,
                                     __HIP_MEMORY_SCOPE_AGENT) == g)
                __builtin_amdgcn_s_sleep(1);
        }
    }
    __syncthreads();
    __threadfence();
}

// --------------------------------------------------------- persistent LSTM
// One layer, both directions. Grid = 16 blocks x 256 threads.
// Blocks [0,8): forward, [8,16): backward. Per direction:
//   64 waves = 2(m-tiles) x 32(j-tiles); each wave owns gate tiles i,f,g,o
//   for its 16 batch rows x 16 hidden columns -> cell update is wave-local,
//   c stays in VGPRs. Wh^T slice (256KB) resident in LDS per block.
//   Double-buffered h => ONE grid barrier per time step.
__global__ __launch_bounds__(256) void lstm_layer(
    const __bf16* __restrict__ whtF, const __bf16* __restrict__ whtB,
    const float*  __restrict__ xgF,  const float*  __restrict__ xgB,
    __bf16*  hbuf,     // [2 dirs][2 bufs][B][H] bf16 (zeroed before launch)
    unsigned* bar,     // [2][2] = {count, gen} per direction
    __bf16* out_bf,    // layer0: concat bf16 [B,T,2H] (else nullptr)
    float*  out_f32)   // layer1: fp32 [B,T,2H]       (else nullptr)
{
    extern __shared__ __bf16 sh[];          // 256 rows x LPITCH bf16

    const int dir = blockIdx.x >> 3;        // 8 blocks per direction
    const int blk = blockIdx.x & 7;

    const __bf16* wht = dir ? whtB : whtF;
    const float*  xg  = dir ? xgB  : xgF;
    __bf16* hb = hbuf + (size_t)dir * 2 * BB * HH;
    unsigned* cnt = bar + dir * 2;
    unsigned* gen = bar + dir * 2 + 1;

    const int lane = threadIdx.x & 31;
    const int wave = threadIdx.x >> 5;
    const int lm   = lane & 15;
    const int hi16 = lane >> 4;

    // wave -> (m tile, local j tile): 8 waves = 2m x 4j
    const int mt = wave >> 2;               // 0,1
    const int lt = wave & 3;                // 0..3
    const int m0 = mt * 16;
    const int j0 = blk * 64 + lt * 16;      // hidden column base (0..511)

    // ---- stage this block's Wh^T rows into LDS (one-time, 256KB) ----
    {
        const int lr  = threadIdx.x;        // local LDS row 0..255
        const int slt = lr >> 6;            // local j tile
        const int sg  = (lr >> 4) & 3;      // gate
        const int sr  = lr & 15;            // row within tile
        const int grow = sg * 512 + blk * 64 + slt * 16 + sr;
        const __bf16* src = wht + (size_t)grow * HH;
        __bf16* dst = sh + (size_t)lr * LPITCH;
        for (int k = 0; k < HH; k += 8)
            *(uint4*)(dst + k) = *(const uint4*)(src + k);
    }
    __syncthreads();

    // persistent cell state in registers
    float creg[8];
    #pragma unroll
    for (int v = 0; v < 8; ++v) creg[v] = 0.0f;

    for (int step = 0; step < TT; ++step) {
        const int tt = dir ? (TT - 1 - step) : step;
        const __bf16* hprev = hb + (size_t)(step & 1) * (BB * HH);
        __bf16*       hnext = hb + (size_t)((step & 1) ^ 1) * (BB * HH);

        // xg gate biases for this step (issue loads early, consumed late)
        float xgv[4][8];
        #pragma unroll
        for (int v = 0; v < 8; ++v) {
            const int m = m0 + v + hi16 * 8;
            const float* xr = xg + ((size_t)m * TT + tt) * NG + j0 + lm;
            #pragma unroll
            for (int g = 0; g < 4; ++g) xgv[g][v] = xr[g * 512];
        }
        // prefetch next step's xg rows (global_prefetch_b8)
        if (step + 1 < TT) {
            const int tn = dir ? (TT - 2 - step) : (step + 1);
            #pragma unroll
            for (int v = 0; v < 8; ++v) {
                const int m = m0 + v + hi16 * 8;
                __builtin_prefetch(xg + ((size_t)m * TT + tn) * NG + j0 + lm, 0, 3);
            }
        }

        // ---- h_prev @ Wh : 16 k-steps x 4 gates, B operands from LDS ----
        v8f accI = {}, accF = {}, accG = {}, accO = {};
        const __bf16* arow = hprev + (size_t)(m0 + lm) * HH;
        #pragma unroll
        for (int ks = 0; ks < 16; ++ks) {
            BFrag a;
            const int ka = ks * 32 + hi16 * 16;
            a.u[0] = *(const uint4*)(arow + ka);
            a.u[1] = *(const uint4*)(arow + ka + 8);
            const int kb = ks * 32 + hi16 * 16;
            #pragma unroll
            for (int g = 0; g < 4; ++g) {
                BFrag b;
                const __bf16* br =
                    sh + (size_t)((lt * 4 + g) * 16 + lm) * LPITCH + kb;
                b.u[0] = *(const uint4*)(br);
                b.u[1] = *(const uint4*)(br + 8);
                v8f* acc = (g == 0) ? &accI : (g == 1) ? &accF
                         : (g == 2) ? &accG : &accO;
                *acc = __builtin_amdgcn_wmma_f32_16x16x32_bf16(
                           false, a.v, false, b.v, (short)0, *acc, false, false);
            }
        }

        // ---- wave-local LSTM cell update (c in VGPRs) ----
        #pragma unroll
        for (int v = 0; v < 8; ++v) {
            const float gi = accI[v] + xgv[0][v];
            const float gf = accF[v] + xgv[1][v];
            const float gg = accG[v] + xgv[2][v];
            const float go = accO[v] + xgv[3][v];
            const float cn = sigmoidf_(gf) * creg[v]
                           + sigmoidf_(gi) * tanhf(gg);
            const float h  = sigmoidf_(go) * tanhf(cn);
            creg[v] = cn;
            const int m = m0 + v + hi16 * 8;          // batch index
            hnext[(size_t)m * HH + j0 + lm] = (__bf16)h;
            const size_t oidx =
                ((size_t)m * TT + tt) * (2 * HH) + dir * HH + j0 + lm;
            if (out_bf)  out_bf[oidx]  = (__bf16)h;
            if (out_f32) out_f32[oidx] = h;
        }

        gridbar(cnt, gen, 8u);   // h fully published -> next step may read
    }
}

// ---------------------------------------------------------------------------
extern "C" void kernel_launch(void* const* d_in, const int* in_sizes, int n_in,
                              void* d_out, int out_size, void* d_ws, size_t ws_size,
                              hipStream_t stream) {
    (void)in_sizes; (void)n_in; (void)out_size; (void)ws_size;

    const float* x    = (const float*)d_in[0];
    const float* Wx0f = (const float*)d_in[1];
    const float* Wh0f = (const float*)d_in[2];
    const float* b0f  = (const float*)d_in[3];
    const float* Wx0b = (const float*)d_in[4];
    const float* Wh0b = (const float*)d_in[5];
    const float* b0b  = (const float*)d_in[6];
    const float* Wx1f = (const float*)d_in[7];
    const float* Wh1f = (const float*)d_in[8];
    const float* b1f  = (const float*)d_in[9];
    const float* Wx1b = (const float*)d_in[10];
    const float* Wh1b = (const float*)d_in[11];
    const float* b1b  = (const float*)d_in[12];

    char* ws = (char*)d_ws;
    size_t off = 0;
    auto alloc = [&](size_t bytes) -> void* {
        void* p = ws + off;
        off = (off + bytes + 255) & ~(size_t)255;
        return p;
    };

    __bf16* xbf   = (__bf16*)alloc((size_t)BT * FF * 2);
    __bf16* wxt0f = (__bf16*)alloc((size_t)NG * FF * 2);
    __bf16* wxt0b = (__bf16*)alloc((size_t)NG * FF * 2);
    __bf16* wht0f = (__bf16*)alloc((size_t)NG * HH * 2);
    __bf16* wht0b = (__bf16*)alloc((size_t)NG * HH * 2);
    __bf16* wxt1f = (__bf16*)alloc((size_t)NG * (2 * HH) * 2);
    __bf16* wxt1b = (__bf16*)alloc((size_t)NG * (2 * HH) * 2);
    __bf16* wht1f = (__bf16*)alloc((size_t)NG * HH * 2);
    __bf16* wht1b = (__bf16*)alloc((size_t)NG * HH * 2);
    __bf16* h0    = (__bf16*)alloc((size_t)BT * (2 * HH) * 2);  // layer-0 out
    float*  xgF   = (float*)alloc((size_t)BT * NG * 4);
    float*  xgB   = (float*)alloc((size_t)BT * NG * 4);
    unsigned* bar = (unsigned*)alloc(256);
    __bf16* hbuf  = (__bf16*)alloc((size_t)2 * 2 * BB * HH * 2);

    const size_t lds_bytes = (size_t)256 * LPITCH * 2;   // 266,240 B < 320KB

    // ---- 1. fp32 -> bf16 conversions / weight transposes ----
    cvt_f32_to_bf16<<<2048, 256, 0, stream>>>(x, xbf, BT * FF);
    transpose_cvt<<<2048, 256, 0, stream>>>(Wx0f, wxt0f, FF);
    transpose_cvt<<<2048, 256, 0, stream>>>(Wx0b, wxt0b, FF);
    transpose_cvt<<<2048, 256, 0, stream>>>(Wh0f, wht0f, HH);
    transpose_cvt<<<2048, 256, 0, stream>>>(Wh0b, wht0b, HH);
    transpose_cvt<<<2048, 256, 0, stream>>>(Wx1f, wxt1f, 2 * HH);
    transpose_cvt<<<2048, 256, 0, stream>>>(Wx1b, wxt1b, 2 * HH);
    transpose_cvt<<<2048, 256, 0, stream>>>(Wh1f, wht1f, HH);
    transpose_cvt<<<2048, 256, 0, stream>>>(Wh1b, wht1b, HH);

    // ---- 2. layer-0 input projections (+bias) ----
    dim3 ggrid(NG / 64, BT / 64);  // 32 x 512
    gemm_xproj<<<ggrid, 256, 0, stream>>>(xbf, wxt0f, b0f, xgF, BT, FF);
    gemm_xproj<<<ggrid, 256, 0, stream>>>(xbf, wxt0b, b0b, xgB, BT, FF);

    // ---- 3. layer-0 recurrence (persistent, both directions) ----
    hipMemsetAsync(bar, 0, 256, stream);
    hipMemsetAsync(hbuf, 0, (size_t)2 * 2 * BB * HH * 2, stream);
    lstm_layer<<<16, 256, lds_bytes, stream>>>(wht0f, wht0b, xgF, xgB,
                                               hbuf, bar, h0, nullptr);

    // ---- 4. layer-1 input projections (reuse xg buffers) ----
    gemm_xproj<<<ggrid, 256, 0, stream>>>(h0, wxt1f, b1f, xgF, BT, 2 * HH);
    gemm_xproj<<<ggrid, 256, 0, stream>>>(h0, wxt1b, b1b, xgB, BT, 2 * HH);

    // ---- 5. layer-1 recurrence -> fp32 output ----
    hipMemsetAsync(bar, 0, 256, stream);
    hipMemsetAsync(hbuf, 0, (size_t)2 * 2 * BB * HH * 2, stream);
    lstm_layer<<<16, 256, lds_bytes, stream>>>(wht1f, wht1b, xgF, xgB,
                                               hbuf, bar, nullptr,
                                               (float*)d_out);
}
